// ContentEncoder_56521769615778
// MI455X (gfx1250) — matrix operand order repaired and checked by reading, not compiled, not run on hardware
//
#include <hip/hip_runtime.h>
#include <hip/hip_bf16.h>

typedef __attribute__((ext_vector_type(16))) _Float16 v16h;
typedef __attribute__((ext_vector_type(8)))  float    v8f;

// Problem shape (matches setup_inputs)
#define BATCH 8
#define NPTS  8192
#define MPTS  2048
#define KNN   16
#define CIN   64
#define C0P   96     // 68 padded to 96 (3 k-tiles of 32)
#define C1    128
#define C2    256
#define GROUPS (BATCH * MPTS)

// ---------------------------------------------------------------------------
// 1) Furthest point sampling: one 1024-thread block per batch (8 pts/thread).
//    Writes fps_points output [B,3,M] and fps_pts ws [B*M,3].
// ---------------------------------------------------------------------------
__global__ __launch_bounds__(1024) void fps_kernel(const float* __restrict__ points,
                                                   float* __restrict__ out_fps,
                                                   float* __restrict__ fps_pts) {
    const int b = blockIdx.x;
    const int t = threadIdx.x;
    const float* pb = points + (size_t)b * 3 * NPTS;

    float px[8], py[8], pz[8], d2[8];
#pragma unroll
    for (int j = 0; j < 8; ++j) {
        int n = j * 1024 + t;
        px[j] = pb[n];
        py[j] = pb[NPTS + n];
        pz[j] = pb[2 * NPTS + n];
        d2[j] = 1e10f;
    }

    __shared__ float sv[32];
    __shared__ int   si[32];
    __shared__ int   scur;

    int cur = 0;
    for (int it = 0; it < MPTS; ++it) {
        float cx = pb[cur], cy = pb[NPTS + cur], cz = pb[2 * NPTS + cur];
        if (t == 0) {
            out_fps[(size_t)b * 3 * MPTS + 0 * MPTS + it] = cx;
            out_fps[(size_t)b * 3 * MPTS + 1 * MPTS + it] = cy;
            out_fps[(size_t)b * 3 * MPTS + 2 * MPTS + it] = cz;
            fps_pts[((size_t)b * MPTS + it) * 3 + 0] = cx;
            fps_pts[((size_t)b * MPTS + it) * 3 + 1] = cy;
            fps_pts[((size_t)b * MPTS + it) * 3 + 2] = cz;
        }
        float bv = -1.0f;
        int   bi = 0;
#pragma unroll
        for (int j = 0; j < 8; ++j) {
            float dx = px[j] - cx, dy = py[j] - cy, dz = pz[j] - cz;
            float d = dx * dx + dy * dy + dz * dz;
            d2[j] = fminf(d2[j], d);
            if (d2[j] > bv) { bv = d2[j]; bi = j * 1024 + t; }
        }
        // wave32 reduce (prefer larger value; ties -> smaller index like argmax)
#pragma unroll
        for (int off = 16; off > 0; off >>= 1) {
            float ov = __shfl_down(bv, off, 32);
            int   oi = __shfl_down(bi, off, 32);
            if (ov > bv || (ov == bv && oi < bi)) { bv = ov; bi = oi; }
        }
        if ((t & 31) == 0) { sv[t >> 5] = bv; si[t >> 5] = bi; }
        __syncthreads();
        if (t < 32) {
            float v = sv[t];
            int   i = si[t];
#pragma unroll
            for (int off = 16; off > 0; off >>= 1) {
                float ov = __shfl_down(v, off, 32);
                int   oi = __shfl_down(i, off, 32);
                if (ov > v || (ov == v && oi < i)) { v = ov; i = oi; }
            }
            if (t == 0) scur = i;
        }
        __syncthreads();
        cur = scur;
    }
}

// ---------------------------------------------------------------------------
// 2) kNN: one wave per anchor. Each lane keeps sorted top-16 of its 256-point
//    slice, then 16 butterfly-min merge rounds (ties -> smaller index).
// ---------------------------------------------------------------------------
__global__ __launch_bounds__(256) void knn_kernel(const float* __restrict__ points,
                                                  const float* __restrict__ fps_pts,
                                                  int* __restrict__ nn_idx) {
    const int lane = threadIdx.x & 31;
    const int w    = threadIdx.x >> 5;
    const int g    = blockIdx.x * 8 + w;       // [0, GROUPS)
    const int b    = g / MPTS;
    const float ax = fps_pts[g * 3 + 0];
    const float ay = fps_pts[g * 3 + 1];
    const float az = fps_pts[g * 3 + 2];
    const float* pb = points + (size_t)b * 3 * NPTS;

    float dist[KNN];
    int   ind[KNN];
#pragma unroll
    for (int j = 0; j < KNN; ++j) { dist[j] = 3.4e38f; ind[j] = 0x7fffffff; }

    for (int i = 0; i < NPTS / 32; ++i) {
        int n = i * 32 + lane;
        float dx = pb[n] - ax, dy = pb[NPTS + n] - ay, dz = pb[2 * NPTS + n] - az;
        float d = dx * dx + dy * dy + dz * dz;
        if (d < dist[KNN - 1]) {
            float cd = d; int ci = n;
#pragma unroll
            for (int j = 0; j < KNN; ++j) {
                if (cd < dist[j]) {
                    float td = dist[j]; int ti = ind[j];
                    dist[j] = cd; ind[j] = ci;
                    cd = td; ci = ti;
                }
            }
        }
    }

    __shared__ float sd[8][32][KNN];
    __shared__ int   sx[8][32][KNN];
#pragma unroll
    for (int j = 0; j < KNN; ++j) { sd[w][lane][j] = dist[j]; sx[w][lane][j] = ind[j]; }
    __syncthreads();

    int head = 0;
    float myd = sd[w][lane][0];
    int   myi = sx[w][lane][0];
    for (int j = 0; j < KNN; ++j) {
        float wd = myd; int wi = myi;
#pragma unroll
        for (int off = 16; off > 0; off >>= 1) {
            float od = __shfl_xor(wd, off, 32);
            int   oi = __shfl_xor(wi, off, 32);
            if (od < wd || (od == wd && oi < wi)) { wd = od; wi = oi; }
        }
        if (lane == 0) nn_idx[g * KNN + j] = wi;
        if (wi == myi && wd == myd) {
            head = (head < KNN - 1) ? head + 1 : head;
            myd = sd[w][lane][head];
            myi = sx[w][lane][head];
        }
    }
}

// ---------------------------------------------------------------------------
// 3) Weight conversion to padded f16 (row-major [O][Cpad] == WMMA B rows)
// ---------------------------------------------------------------------------
__global__ void pack_w_kernel(const float* __restrict__ W1, const float* __restrict__ W2,
                              _Float16* __restrict__ W1p, _Float16* __restrict__ W2p) {
    int i = blockIdx.x * blockDim.x + threadIdx.x;
    const int n1 = C1 * C0P;
    if (i < n1) {
        int o = i / C0P, c = i % C0P;
        W1p[i] = (c < 68) ? (_Float16)W1[o * 68 + c] : (_Float16)0.0f;
    } else if (i < n1 + C2 * C1) {
        int j = i - n1;
        W2p[j] = (_Float16)W2[j];
    }
}

// ---------------------------------------------------------------------------
// WMMA tile loaders (16-bit, 16x16x32 shapes, wave32 layouts per CDNA5 ISA)
// A 16x32: lanes 0-15 hold M=lane, halves K=[0..7,16..23]; lanes 16-31 K=[8..15,24..31]
// B 32x16: lanes 0-15 hold N=lane, halves K=[0..7,16..23]; lanes 16-31 K=[8..15,24..31]
// ---------------------------------------------------------------------------
__device__ inline v16h load_a_tile(const _Float16* p, int stride_h, int row, int kbase, int hi) {
    const _Float16* q = p + row * stride_h + kbase + (hi ? 8 : 0);
    union { uint4 u[2]; v16h v; } t;
    t.u[0] = *(const uint4*)(q);
    t.u[1] = *(const uint4*)(q + 16);
    return t.v;
}
__device__ inline v16h load_b_tile(const _Float16* wgt, int stride_h, int ncol, int kbase, int hi) {
    const _Float16* q = wgt + ncol * stride_h + kbase + (hi ? 8 : 0);
    union { uint4 u[2]; v16h v; } t;
    t.u[0] = *(const uint4*)(q);
    t.u[1] = *(const uint4*)(q + 16);
    return t.v;
}

// ---------------------------------------------------------------------------
// 4) Fused gather -> GEMM1(relu) -> GEMM2(relu) -> maxpool over k.
//    One wave per FPS point; weights staged in LDS, shared by 8 waves.
//    LDS: 24KB (W1) + 64KB (W2) + 8*4KB (x/h1 reused) = 120KB / block.
// ---------------------------------------------------------------------------
__global__ __launch_bounds__(256) void mlp_kernel(const float* __restrict__ points,
                                                  const float* __restrict__ feats,
                                                  const float* __restrict__ b1,
                                                  const float* __restrict__ b2,
                                                  const _Float16* __restrict__ W1p,
                                                  const _Float16* __restrict__ W2p,
                                                  const float* __restrict__ fps_pts,
                                                  const int* __restrict__ nn_idx,
                                                  float* __restrict__ out) {
    __shared__ __attribute__((aligned(32))) _Float16 w1s[C1 * C0P];     // 24 KB
    __shared__ __attribute__((aligned(32))) _Float16 w2s[C2 * C1];      // 64 KB
    __shared__ __attribute__((aligned(32))) _Float16 hb[8][16 * C1];    // 32 KB, x then h1

    const int lane = threadIdx.x & 31;
    const int w    = threadIdx.x >> 5;
    const int g    = blockIdx.x * 8 + w;
    const int b    = g / MPTS;
    const int m    = g % MPTS;
    const int row  = lane & 15;
    const int hi   = lane >> 4;

    // ---- stage weights into LDS (whole block cooperates) ----
    {
        uint4* d1 = (uint4*)w1s; const uint4* s1 = (const uint4*)W1p;
        uint4* d2 = (uint4*)w2s; const uint4* s2 = (const uint4*)W2p;
        const int n1 = C1 * C0P / 8;   // 1536 uint4
        const int n2 = C2 * C1 / 8;    // 4096 uint4
        for (int i = threadIdx.x; i < n1; i += 256) d1[i] = s1[i];
        for (int i = threadIdx.x; i < n2; i += 256) d2[i] = s2[i];
    }

    // ---- gather: build x[row][c] = {rela.xyz, dist, feats[0..63], pad} ----
    _Float16* xb = &hb[w][0];                  // used with stride C0P
    const int nn = nn_idx[g * KNN + row];
    const float fx = fps_pts[g * 3 + 0], fy = fps_pts[g * 3 + 1], fz = fps_pts[g * 3 + 2];
    const float* pb = points + (size_t)b * 3 * NPTS;
    const float rx = pb[nn] - fx;
    const float ry = pb[NPTS + nn] - fy;
    const float rz = pb[2 * NPTS + nn] - fz;
    const float dd = sqrtf(rx * rx + ry * ry + rz * rz + 1e-12f);
    _Float16* xr = xb + row * C0P;
    const float* fb = feats + (size_t)b * CIN * NPTS;
    if (hi == 0) {
        xr[0] = (_Float16)rx; xr[1] = (_Float16)ry; xr[2] = (_Float16)rz; xr[3] = (_Float16)dd;
#pragma unroll
        for (int c = 4; c < 48; ++c) xr[c] = (_Float16)fb[(size_t)(c - 4) * NPTS + nn];
    } else {
#pragma unroll
        for (int c = 48; c < 68; ++c) xr[c] = (_Float16)fb[(size_t)(c - 4) * NPTS + nn];
#pragma unroll
        for (int c = 68; c < C0P; ++c) xr[c] = (_Float16)0.0f;
    }
    __syncthreads();

    // ---- load A tiles for GEMM1, then free the x region for h1 reuse ----
    v16h a1[3];
#pragma unroll
    for (int kt = 0; kt < 3; ++kt) a1[kt] = load_a_tile(xb, C0P, row, kt * 32, hi);
    __syncthreads();   // all x reads complete before h1 overwrites the region

    // ---- GEMM1: h1[16x128] = relu(x[16x96] * W1^T + b1) ----
    _Float16* h1 = &hb[w][0];                  // stride C1
#pragma unroll
    for (int nt = 0; nt < C1 / 16; ++nt) {
        v8f acc = {};
#pragma unroll
        for (int kt = 0; kt < 3; ++kt) {
            v16h bt = load_b_tile(w1s, C0P, nt * 16 + row, kt * 32, hi);
            acc = __builtin_amdgcn_wmma_f32_16x16x32_f16(false, a1[kt], false, bt,
                                                         (short)0, acc, false, false);
        }
        const float bias = b1[nt * 16 + row];
        const int rbase = hi ? 8 : 0;
#pragma unroll
        for (int v = 0; v < 8; ++v) {
            float hv = fmaxf(acc[v] + bias, 0.0f);
            h1[(rbase + v) * C1 + nt * 16 + row] = (_Float16)hv;
        }
    }
    __syncthreads();   // h1 writes visible to all lanes

    // ---- GEMM2 + relu + maxpool over the 16 rows (k dimension) ----
    v16h a2[4];
#pragma unroll
    for (int kt = 0; kt < 4; ++kt) a2[kt] = load_a_tile(h1, C1, row, kt * 32, hi);
    float* ob = out + (size_t)b * C2 * MPTS + m;
#pragma unroll
    for (int nt = 0; nt < C2 / 16; ++nt) {
        v8f acc = {};
#pragma unroll
        for (int kt = 0; kt < 4; ++kt) {
            v16h bt = load_b_tile(w2s, C1, nt * 16 + row, kt * 32, hi);
            acc = __builtin_amdgcn_wmma_f32_16x16x32_f16(false, a2[kt], false, bt,
                                                         (short)0, acc, false, false);
        }
        const float bias = b2[nt * 16 + row];
        float mx = 0.0f;
#pragma unroll
        for (int v = 0; v < 8; ++v) mx = fmaxf(mx, fmaxf(acc[v] + bias, 0.0f));
        mx = fmaxf(mx, __shfl_xor(mx, 16, 32));   // combine rows 0-7 with 8-15
        if (hi == 0) ob[(size_t)(nt * 16 + row) * MPTS] = mx;
    }
}

// ---------------------------------------------------------------------------
extern "C" void kernel_launch(void* const* d_in, const int* in_sizes, int n_in,
                              void* d_out, int out_size, void* d_ws, size_t ws_size,
                              hipStream_t stream) {
    const float* points = (const float*)d_in[0];
    const float* feats  = (const float*)d_in[1];
    const float* W1     = (const float*)d_in[2];
    const float* b1     = (const float*)d_in[3];
    const float* W2     = (const float*)d_in[4];
    const float* b2     = (const float*)d_in[5];

    float* out_fps = (float*)d_out;                          // [B,3,M]
    float* out_mlp = out_fps + (size_t)BATCH * 3 * MPTS;     // [B,256,M]

    // workspace layout
    float*    fps_pts = (float*)d_ws;                        // GROUPS*3 f32
    int*      nn_idx  = (int*)(fps_pts + (size_t)GROUPS * 3);// GROUPS*16 i32
    _Float16* W1p     = (_Float16*)(nn_idx + (size_t)GROUPS * KNN); // 128*96
    _Float16* W2p     = W1p + C1 * C0P;                      // 256*128

    fps_kernel<<<BATCH, 1024, 0, stream>>>(points, out_fps, fps_pts);
    knn_kernel<<<GROUPS / 8, 256, 0, stream>>>(points, fps_pts, nn_idx);
    {
        const int tot = C1 * C0P + C2 * C1;
        pack_w_kernel<<<(tot + 255) / 256, 256, 0, stream>>>(W1, W2, W1p, W2p);
    }
    mlp_kernel<<<GROUPS / 8, 256, 0, stream>>>(points, feats, b1, b2, W1p, W2p,
                                               fps_pts, nn_idx, out_mlp);
}
